// Linear2Bit_30691836297883
// MI455X (gfx1250) — matrix operand order repaired and checked by reading, not compiled, not run on hardware
//
#include <hip/hip_runtime.h>

typedef __attribute__((ext_vector_type(16))) _Float16 v16h;
typedef __attribute__((ext_vector_type(8)))  _Float16 v8h;
typedef __attribute__((ext_vector_type(8)))  float    v8f;

#define IN_F  4096
#define OUT_F 4096
#define BM 128
#define BN 128
#define BK 128
#define LDSK 136   // padded LDS row stride in f16 elements (272 bytes, 16B-aligned)

// ---------------------------------------------------------------------------
// Kernel 1: one-shot dequantization of the whole 2-bit weight matrix to f16.
// Output layout [OUT_F][IN_F] row-major; since groups are contiguous along K,
// dst address is simply wdq + int_index*4.  Reads 16MB, writes 32MB (one pass).
// ---------------------------------------------------------------------------
__global__ __launch_bounds__(256)
void q2_dequant_kernel(const int* __restrict__ wq,
                       const float* __restrict__ wnorm,
                       _Float16* __restrict__ wdq)
{
  int t = blockIdx.x * 256 + threadIdx.x;   // packed-int index
  int g = t >> 5;                           // quant group (32 ints per group)
  float nm = wnorm[g];
  float s  = nm * (2.0f / 3.0f);
  int q = wq[t];
  union { _Float16 h[4]; uint2 u; } p;
  #pragma unroll
  for (int b = 0; b < 4; ++b) {
    float u = (float)((q >> (2 * b)) & 3);
    p.h[b] = (_Float16)__builtin_fmaf(u, s, -nm);   // u*(2/3)*norm - norm
  }
  *(uint2*)(wdq + (size_t)t * 4) = p.u;
}

// ---------------------------------------------------------------------------
// Kernel 2: GEMM on pre-dequantized f16 weights (L2-resident, 32MB << 192MB).
// W tile staged via CDNA5 async global->LDS copy (ASYNCcnt path), X tile
// converted fp32->f16 through VGPRs.
// ---------------------------------------------------------------------------
__global__ __launch_bounds__(256)
void q2lin_wmma_pre(const float* __restrict__ x,
                    const _Float16* __restrict__ wdq,
                    const float* __restrict__ bias,
                    float* __restrict__ out)
{
  __shared__ __align__(16) _Float16 Xs[BM * LDSK];
  __shared__ __align__(16) _Float16 Ws[BN * LDSK];

  const int tid   = threadIdx.x;
  const int lane  = tid & 31;
  const int wave  = tid >> 5;
  const int waveM = wave >> 2;
  const int waveN = wave & 3;
  const int r16   = lane & 15;
  const int kh    = lane >> 4;

  const int  n0 = blockIdx.x * BN;
  const long m0 = (long)blockIdx.y * BM;

  v8f acc[4][2];
  #pragma unroll
  for (int i = 0; i < 4; ++i)
    #pragma unroll
    for (int j = 0; j < 2; ++j)
      acc[i][j] = {};

  // per-thread W-copy geometry: 2 threads per n-row, 8 x 16B chunks each
  const int wrow = tid >> 1;
  const int wh   = (tid & 1) * 128;              // byte offset of 64-K half
  const unsigned ldsW = ((unsigned)(size_t)&Ws[0]) + (unsigned)(wrow * (LDSK * 2) + wh);

  for (int kc = 0; kc < IN_F; kc += BK) {
    // ---- async-copy W tile (f16, 128x128 = 32KB) straight into LDS ----
    {
      unsigned goff = (unsigned)((n0 + wrow) * (IN_F * 2) + kc * 2 + wh);
      #pragma unroll
      for (int c = 0; c < 8; ++c) {
        asm volatile("global_load_async_to_lds_b128 %0, %1, %2 offset:%3"
                     :: "v"(ldsW), "v"(goff), "s"(wdq), "i"(c * 16)
                     : "memory");
      }
    }

    // ---- stage X tile: 128x128 fp32 -> f16 into LDS (coalesced float4) ----
    #pragma unroll
    for (int i = 0; i < 16; ++i) {
      int flat = i * 256 + tid;
      int row  = flat >> 5;
      int c4   = flat & 31;
      const float4 v = *(const float4*)(x + (m0 + row) * IN_F + kc + c4 * 4);
      union { _Float16 h[4]; uint2 u; } p;
      p.h[0] = (_Float16)v.x; p.h[1] = (_Float16)v.y;
      p.h[2] = (_Float16)v.z; p.h[3] = (_Float16)v.w;
      *(uint2*)(&Xs[row * LDSK + c4 * 4]) = p.u;
      if ((i & 3) == 0 && kc + BK < IN_F)
        __builtin_prefetch(x + (m0 + row) * IN_F + kc + BK + c4 * 4, 0, 0);
    }

    asm volatile("s_wait_asynccnt 0x0" ::: "memory");
    __syncthreads();

    // ---- 4 WMMA K-steps of 32 over the staged 128-K chunk ----
    #pragma unroll
    for (int k0 = 0; k0 < BK; k0 += 32) {
      v16h a[4], b[2];
      #pragma unroll
      for (int i = 0; i < 4; ++i) {
        const _Float16* p = &Xs[(waveM * 64 + i * 16 + r16) * LDSK + k0 + kh * 16];
        v8h lo = *(const v8h*)p;
        v8h hi = *(const v8h*)(p + 8);
        a[i] = __builtin_shufflevector(lo, hi, 0,1,2,3,4,5,6,7,8,9,10,11,12,13,14,15);
      }
      #pragma unroll
      for (int j = 0; j < 2; ++j) {
        const _Float16* p = &Ws[(waveN * 32 + j * 16 + r16) * LDSK + k0 + kh * 16];
        v8h lo = *(const v8h*)p;
        v8h hi = *(const v8h*)(p + 8);
        b[j] = __builtin_shufflevector(lo, hi, 0,1,2,3,4,5,6,7,8,9,10,11,12,13,14,15);
      }
      #pragma unroll
      for (int i = 0; i < 4; ++i)
        #pragma unroll
        for (int j = 0; j < 2; ++j)
          acc[i][j] = __builtin_amdgcn_wmma_f32_16x16x32_f16(
              false, a[i], false, b[j], (short)0, acc[i][j], false, false);
    }
    __syncthreads();
  }

  #pragma unroll
  for (int j = 0; j < 2; ++j) {
    int n = n0 + waveN * 32 + j * 16 + r16;
    float bv = bias[n];
    #pragma unroll
    for (int i = 0; i < 4; ++i) {
      long mb = m0 + waveM * 64 + i * 16 + kh * 8;
      #pragma unroll
      for (int r = 0; r < 8; ++r)
        out[(mb + r) * OUT_F + n] = acc[i][j][r] + bv;
    }
  }
}

// ---------------------------------------------------------------------------
// Fallback: fused dequant + GEMM (used only if d_ws is too small).
// ---------------------------------------------------------------------------
__global__ __launch_bounds__(256)
void q2lin_wmma_fused(const float* __restrict__ x,
                      const int*   __restrict__ wq,
                      const float* __restrict__ wnorm,
                      const float* __restrict__ bias,
                      float* __restrict__ out)
{
  __shared__ __align__(16) _Float16 Xs[BM * LDSK];
  __shared__ __align__(16) _Float16 Ws[BN * LDSK];

  const int tid   = threadIdx.x;
  const int lane  = tid & 31;
  const int wave  = tid >> 5;
  const int waveM = wave >> 2;
  const int waveN = wave & 3;
  const int r16   = lane & 15;
  const int kh    = lane >> 4;

  const int  n0 = blockIdx.x * BN;
  const long m0 = (long)blockIdx.y * BM;

  v8f acc[4][2];
  #pragma unroll
  for (int i = 0; i < 4; ++i)
    #pragma unroll
    for (int j = 0; j < 2; ++j)
      acc[i][j] = {};

  const int nrow = tid >> 1;
  const int half = tid & 1;

  for (int kc = 0; kc < IN_F; kc += BK) {
    #pragma unroll
    for (int i = 0; i < 16; ++i) {
      int flat = i * 256 + tid;
      int row  = flat >> 5;
      int c4   = flat & 31;
      const float4 v = *(const float4*)(x + (m0 + row) * IN_F + kc + c4 * 4);
      union { _Float16 h[4]; uint2 u; } p;
      p.h[0] = (_Float16)v.x; p.h[1] = (_Float16)v.y;
      p.h[2] = (_Float16)v.z; p.h[3] = (_Float16)v.w;
      *(uint2*)(&Xs[row * LDSK + c4 * 4]) = p.u;
    }
    {
      int g = (n0 + nrow) * (IN_F / 128) + (kc >> 7);
      const int* qp = wq + (long)g * 32 + half * 16;
      float nm = wnorm[g];
      float s  = nm * (2.0f / 3.0f);
      _Float16* dst = &Ws[nrow * LDSK + half * 64];
      #pragma unroll
      for (int c = 0; c < 16; ++c) {
        int q = qp[c];
        union { _Float16 h[4]; uint2 u; } p;
        #pragma unroll
        for (int b = 0; b < 4; ++b) {
          float u = (float)((q >> (2 * b)) & 3);
          p.h[b] = (_Float16)__builtin_fmaf(u, s, -nm);
        }
        *(uint2*)(dst + c * 4) = p.u;
      }
    }
    __syncthreads();

    #pragma unroll
    for (int k0 = 0; k0 < BK; k0 += 32) {
      v16h a[4], b[2];
      #pragma unroll
      for (int i = 0; i < 4; ++i) {
        const _Float16* p = &Xs[(waveM * 64 + i * 16 + r16) * LDSK + k0 + kh * 16];
        v8h lo = *(const v8h*)p;
        v8h hi = *(const v8h*)(p + 8);
        a[i] = __builtin_shufflevector(lo, hi, 0,1,2,3,4,5,6,7,8,9,10,11,12,13,14,15);
      }
      #pragma unroll
      for (int j = 0; j < 2; ++j) {
        const _Float16* p = &Ws[(waveN * 32 + j * 16 + r16) * LDSK + k0 + kh * 16];
        v8h lo = *(const v8h*)p;
        v8h hi = *(const v8h*)(p + 8);
        b[j] = __builtin_shufflevector(lo, hi, 0,1,2,3,4,5,6,7,8,9,10,11,12,13,14,15);
      }
      #pragma unroll
      for (int i = 0; i < 4; ++i)
        #pragma unroll
        for (int j = 0; j < 2; ++j)
          acc[i][j] = __builtin_amdgcn_wmma_f32_16x16x32_f16(
              false, a[i], false, b[j], (short)0, acc[i][j], false, false);
    }
    __syncthreads();
  }

  #pragma unroll
  for (int j = 0; j < 2; ++j) {
    int n = n0 + waveN * 32 + j * 16 + r16;
    float bv = bias[n];
    #pragma unroll
    for (int i = 0; i < 4; ++i) {
      long mb = m0 + waveM * 64 + i * 16 + kh * 8;
      #pragma unroll
      for (int r = 0; r < 8; ++r)
        out[(mb + r) * OUT_F + n] = acc[i][j][r] + bv;
    }
  }
}

extern "C" void kernel_launch(void* const* d_in, const int* in_sizes, int n_in,
                              void* d_out, int out_size, void* d_ws, size_t ws_size,
                              hipStream_t stream) {
  (void)n_in; (void)out_size;
  const float* x     = (const float*)d_in[0];
  const int*   wq    = (const int*)d_in[1];
  const float* wnorm = (const float*)d_in[2];
  const float* bias  = (const float*)d_in[3];
  float* out = (float*)d_out;

  int M = in_sizes[0] / IN_F;            // 16384
  dim3 grid(OUT_F / BN, M / BM);         // (32, 128)

  const size_t need = (size_t)OUT_F * IN_F * sizeof(_Float16);   // 32 MB
  if (ws_size >= need) {
    _Float16* wdq = (_Float16*)d_ws;
    int n_ints = OUT_F * IN_F / 4;       // 4,194,304
    q2_dequant_kernel<<<n_ints / 256, 256, 0, stream>>>(wq, wnorm, wdq);
    q2lin_wmma_pre<<<grid, 256, 0, stream>>>(x, wdq, bias, out);
  } else {
    q2lin_wmma_fused<<<grid, 256, 0, stream>>>(x, wq, wnorm, bias, out);
  }
}
